// DeformEncoder_2327872274598
// MI455X (gfx1250) — compile-verified, matrix operand order checked by reading, not run on hardware
//
#include <hip/hip_runtime.h>
#include <math.h>

// ---------------------------------------------------------------------------
// Deformable-DETR encoder forward for MI455X (gfx1250, wave32, WMMA).
// GEMMs: v_wmma_f32_16x16x32_bf16, double-buffered LDS tiles staged with
// GLOBAL_LOAD_ASYNC_TO_LDS_B128 (ASYNCcnt) when the toolchain exposes it.
// ---------------------------------------------------------------------------

typedef __bf16 bf16_t;
typedef __attribute__((ext_vector_type(16))) __bf16 v16bf;
typedef __attribute__((ext_vector_type(8)))  float  v8f;
typedef int v4i_vs __attribute__((vector_size(16)));   // matches builtin's V4i

#define NL    6
#define CDIM  256
#define NH    8
#define HD    32
#define NP    4
#define DF    1024
#define HWN   4096          // 64*64
#define TOK   32768         // B(8) * HWN
#define GSZ   64            // H == W == 64

// ---- async global->LDS staging (gfx1250), with safe fallback --------------
#if defined(__gfx1250__) && __has_builtin(__builtin_amdgcn_global_load_async_to_lds_b128)
#define ASYNC_COPY_B128(gp, sp)                                              \
  __builtin_amdgcn_global_load_async_to_lds_b128(                            \
      (__attribute__((address_space(1))) v4i_vs*)(gp),                       \
      (__attribute__((address_space(3))) v4i_vs*)(sp), 0, 0)
#if __has_builtin(__builtin_amdgcn_s_wait_asynccnt)
#define ASYNC_WAIT() __builtin_amdgcn_s_wait_asynccnt(0)
#else
#define ASYNC_WAIT() asm volatile("s_wait_asynccnt 0x0" ::: "memory")
#endif
#else
#define ASYNC_COPY_B128(gp, sp) \
  do { *(uint4*)(sp) = *(const uint4*)(gp); } while (0)
#define ASYNC_WAIT() do { } while (0)
#endif

// LDS tile row stride: 40 bf16 = 80B = 20 dwords -> 16 fragment rows map to
// 16 distinct 4-dword groups mod 64 banks (conflict-free ds_load_b128).
#define LDA 40

__device__ __forceinline__ v8f wmma_bf16(v16bf a, v16bf b, v8f c) {
  return __builtin_amdgcn_wmma_f32_16x16x32_bf16(false, a, false, b, (short)0,
                                                 c, false, false);
}

// ---------------------------------------------------------------------------
// WMMA fragment load from row-major bf16 [rows][LDA] LDS storage.
// 16-bit A/B layout (ISA 7.12.2): lane 0-15 -> row = lane, elems 0..7 = K 0..7,
// elems 8..15 = K 16..23 ; lane 16-31 -> row = lane-16, K 8..15 / 24..31.
// ---------------------------------------------------------------------------
__device__ __forceinline__ v16bf load_frag(const bf16_t* base, int ld) {
  const int lane = threadIdx.x & 31;
  const int r    = lane & 15;
  const int kb   = (lane < 16) ? 0 : 8;
  union { v16bf v; uint4 q[2]; } u;
  const bf16_t* p = base + r * ld;
  u.q[0] = *(const uint4*)(p + kb);        // 8 bf16 = 16 bytes
  u.q[1] = *(const uint4*)(p + kb + 16);   // 8 bf16 = 16 bytes
  return u.v;
}

// ---------------------------------------------------------------------------
// Generic GEMM:  C[M][N] = A[M][K](bf16) * W[N][K](bf16)^T + bias[N]
// 256 threads = 8 waves arranged WM_WAVES x WN_WAVES; wave tile 32 x WNT
// (2 x WNT/16 WMMA fragments). K staged 32-wide through double-buffered LDS
// via async global->LDS b128 loads. Per-thread chunk pointers are hoisted out
// of the K loop (K-invariant) so each issue is straight-line async loads.
// Requires M%BM==0, N%BN==0, K%32==0.
// ---------------------------------------------------------------------------
template <int WM_WAVES, int WN_WAVES, int WNT>
__global__ __launch_bounds__(256)
void k_gemm(const bf16_t* __restrict__ A, const bf16_t* __restrict__ W,
            const float* __restrict__ bias, float* __restrict__ Cf,
            bf16_t* __restrict__ Cb, int M, int N, int K, int relu)
{
  constexpr int BM = WM_WAVES * 32;
  constexpr int BN = WN_WAVES * WNT;
  constexpr int NFRAG  = WNT / 16;
  constexpr int A_IT   = (BM * 4) / 256;   // 16B chunks per thread, A tile
  constexpr int B_FULL = (BN * 4) / 256;   // full 256-thread rounds, B tile
  constexpr int B_REM  = (BN * 4) % 256;   // partial round (first B_REM threads)

  __shared__ __align__(16) bf16_t sA[2][BM * LDA];
  __shared__ __align__(16) bf16_t sB[2][BN * LDA];

  const int tid   = threadIdx.x;
  const int lane  = tid & 31;
  const int wave  = tid >> 5;
  const int waveM = wave % WM_WAVES;
  const int waveN = wave / WM_WAVES;
  const int mBase = blockIdx.x * BM;
  const int nBase = blockIdx.y * BN;

  v8f acc[2][NFRAG];
#pragma unroll
  for (int i = 0; i < 2; ++i)
#pragma unroll
    for (int j = 0; j < NFRAG; ++j) acc[i][j] = (v8f){};

  // ---- K-invariant per-thread staging addresses (chunk = 16B = 8 bf16) ----
  const int cr = tid >> 2;            // chunk row within first round
  const int cq = (tid & 3) * 8;       // bf16 offset of 16B chunk within row

  const bf16_t* gAp[A_IT];
  int sAo[A_IT];
#pragma unroll
  for (int i = 0; i < A_IT; ++i) {
    const int r = cr + i * 64;
    gAp[i] = A + (size_t)(mBase + r) * K + cq;
    sAo[i] = r * LDA + cq;
  }

  const bf16_t* gBp[B_FULL ? B_FULL : 1];
  int sBo[B_FULL ? B_FULL : 1];
#pragma unroll
  for (int i = 0; i < B_FULL; ++i) {
    const int r = cr + i * 64;
    gBp[i] = W + (size_t)(nBase + r) * K + cq;
    sBo[i] = r * LDA + cq;
  }
  const bf16_t* gBr = W;
  int sBr = 0;
  if constexpr (B_REM != 0) {
    const int r = cr + B_FULL * 64;   // only meaningful for tid < B_REM
    gBr = W + (size_t)(nBase + (tid < B_REM ? r : 0)) * K + cq;
    sBr = r * LDA + cq;
  }

  // straight-line async stage of one 32-wide K slice into LDS buffer `buf`
  auto issue = [&](int k0, int buf) {
#pragma unroll
    for (int i = 0; i < A_IT; ++i)
      ASYNC_COPY_B128(gAp[i] + k0, &sA[buf][sAo[i]]);
#pragma unroll
    for (int i = 0; i < B_FULL; ++i)
      ASYNC_COPY_B128(gBp[i] + k0, &sB[buf][sBo[i]]);
    if constexpr (B_REM != 0) {
      if (tid < B_REM)
        ASYNC_COPY_B128(gBr + k0, &sB[buf][sBr]);
    }
  };

  issue(0, 0);
  int buf = 0;
  for (int k0 = 0; k0 < K; k0 += 32) {
    ASYNC_WAIT();        // my async loads into `buf` complete
    __syncthreads();     // everyone's complete; prior readers of buf^1 done
    if (k0 + 32 < K) issue(k0 + 32, buf ^ 1);

    const bf16_t* aB = &sA[buf][(waveM * 32) * LDA];
    const bf16_t* bB = &sB[buf][(waveN * WNT) * LDA];
    v16bf af0 = load_frag(aB, LDA);
    v16bf af1 = load_frag(aB + 16 * LDA, LDA);
#pragma unroll
    for (int j = 0; j < NFRAG; ++j) {
      v16bf bfj = load_frag(bB + j * 16 * LDA, LDA);
      acc[0][j] = wmma_bf16(af0, bfj, acc[0][j]);
      acc[1][j] = wmma_bf16(af1, bfj, acc[1][j]);
    }
    buf ^= 1;
  }

  // D layout (ISA 7.12.2): VGPR r, lanes 0-15 -> M=r, N=lane ; lanes16-31 -> M=r+8.
  const int mw    = mBase + waveM * 32;
  const int nw    = nBase + waveN * WNT;
  const int nlane = lane & 15;
  const int mhi   = (lane >> 4) << 3;
#pragma unroll
  for (int i = 0; i < 2; ++i)
#pragma unroll
    for (int j = 0; j < NFRAG; ++j) {
      const int n = nw + j * 16 + nlane;
      const float bval = bias ? bias[n] : 0.0f;
#pragma unroll
      for (int r = 0; r < 8; ++r) {
        const int m = mw + i * 16 + mhi + r;
        float v = acc[i][j][r] + bval;
        if (relu) v = fmaxf(v, 0.0f);
        if (Cb) Cb[(size_t)m * N + n] = (bf16_t)v;
        else    Cf[(size_t)m * N + n] = v;
      }
    }
}

// ---------------------------------------------------------------------------
// Sine positional encoding table pos[HWN][CDIM]   (DETR sine, normalize=True)
// ---------------------------------------------------------------------------
__global__ void k_pos(float* __restrict__ pos)
{
  const int t = blockIdx.x * blockDim.x + threadIdx.x;   // HWN * 128 threads
  if (t >= HWN * 128) return;
  const int hw = t >> 7, k2 = t & 127;
  const int y = hw >> 6, x = hw & 63;
  const int half = (k2 >= 64) ? 1 : 0;
  const int k = k2 - half * 64;
  float v = half ? (float)(x + 1) : (float)(y + 1);
  v = v / (64.0f + 1e-6f) * 6.283185307179586f;
  const float freq = powf(10000.0f, (float)k / 64.0f);
  const float p = v / freq;
  float* dst = pos + (size_t)hw * CDIM + half * 128 + 2 * k;
  dst[0] = sinf(p);
  dst[1] = cosf(p);
}

// NCHW -> (token, C)
__global__ void k_transpose_in(const float* __restrict__ x, float* __restrict__ srcF)
{
  const int t = blockIdx.x * blockDim.x + threadIdx.x;   // TOK*CDIM
  if (t >= TOK * CDIM) return;
  const int n = t >> 8, c = t & 255;
  const int b = n >> 12, hw = n & 4095;
  srcF[t] = x[((size_t)b * CDIM + c) * HWN + hw];
}

// (token, C) -> NCHW
__global__ void k_transpose_out(const float* __restrict__ srcF, float* __restrict__ out)
{
  const int t = blockIdx.x * blockDim.x + threadIdx.x;   // TOK*CDIM
  if (t >= TOK * CDIM) return;
  const int n = t >> 8, c = t & 255;
  const int b = n >> 12, hw = n & 4095;
  out[((size_t)b * CDIM + c) * HWN + hw] = srcF[t];
}

// srcB = bf16(src);  qB = bf16(src + pos)
__global__ void k_prep(const float* __restrict__ srcF, const float* __restrict__ pos,
                       bf16_t* __restrict__ srcB, bf16_t* __restrict__ qB)
{
  const int t = blockIdx.x * blockDim.x + threadIdx.x;   // TOK*CDIM
  if (t >= TOK * CDIM) return;
  const int c = t & 255, hw = (t >> 8) & 4095;
  const float s = srcF[t];
  srcB[t] = (bf16_t)s;
  qB[t]   = (bf16_t)(s + pos[(size_t)hw * CDIM + c]);
}

// ---------------------------------------------------------------------------
// Fused residual + LayerNorm (row = token, one wave32 per row).
// srcF = LN(srcF + addF)*w + b ; srcB = bf16(srcF) ; optionally qB = bf16(srcF+pos)
// ---------------------------------------------------------------------------
__global__ __launch_bounds__(256)
void k_add_ln(float* __restrict__ srcF, const float* __restrict__ addF,
              const float* __restrict__ w, const float* __restrict__ b,
              bf16_t* __restrict__ srcB, const float* __restrict__ pos,
              bf16_t* __restrict__ qB)
{
  const int row  = blockIdx.x * 8 + (threadIdx.x >> 5);
  const int lane = threadIdx.x & 31;
  float* rs = srcF + (size_t)row * CDIM;
  const float* ra = addF + (size_t)row * CDIM;

  float vals[8];
  float s = 0.0f, sq = 0.0f;
#pragma unroll
  for (int i = 0; i < 8; ++i) {
    const float xv = rs[lane + i * 32] + ra[lane + i * 32];
    vals[i] = xv; s += xv; sq += xv * xv;
  }
#pragma unroll
  for (int off = 16; off > 0; off >>= 1) {
    s  += __shfl_xor(s,  off, 32);
    sq += __shfl_xor(sq, off, 32);
  }
  const float mean = s * (1.0f / 256.0f);
  const float var  = sq * (1.0f / 256.0f) - mean * mean;
  const float inv  = rsqrtf(var + 1e-5f);
  const float* prow = pos ? (pos + (size_t)(row & 4095) * CDIM) : (const float*)0;

#pragma unroll
  for (int i = 0; i < 8; ++i) {
    const int c = lane + i * 32;
    const float yv = (vals[i] - mean) * inv * w[c] + b[c];
    rs[c] = yv;
    srcB[(size_t)row * CDIM + c] = (bf16_t)yv;
    if (qB) qB[(size_t)row * CDIM + c] = (bf16_t)(yv + prow[c]);
  }
}

// ---------------------------------------------------------------------------
// Fused softmax(4) + bilinear deformable gather.
// One wave32 per (token, head); lane = channel within head (hd=32) so every
// corner read is one coalesced 128B burst per wave.
// oa row layout: [0:64) = offsets (head,point,xy), [64:96) = attn logits.
// ---------------------------------------------------------------------------
__global__ __launch_bounds__(256)
void k_deform_gather(const float* __restrict__ vF, const float* __restrict__ oa,
                     bf16_t* __restrict__ outB)
{
  const int wid  = blockIdx.x * 8 + (threadIdx.x >> 5);   // TOK*NH waves
  const int lane = threadIdx.x & 31;
  const int head = wid & 7;
  const int n    = wid >> 3;
  const int bidx = n >> 12;
  const int hw   = n & 4095;
  const int yy   = hw >> 6, xx = hw & 63;

  const float* oan = oa + (size_t)n * 96;

  // softmax over the 4 sampling points
  float lg[NP];
#pragma unroll
  for (int p = 0; p < NP; ++p) lg[p] = oan[64 + head * 4 + p];
  float mx = lg[0];
#pragma unroll
  for (int p = 1; p < NP; ++p) mx = fmaxf(mx, lg[p]);
  float esum = 0.0f;
#pragma unroll
  for (int p = 0; p < NP; ++p) { lg[p] = expf(lg[p] - mx); esum += lg[p]; }
  const float rsum = 1.0f / esum;

  const float refx = ((float)xx + 0.5f) * (1.0f / 64.0f);
  const float refy = ((float)yy + 0.5f) * (1.0f / 64.0f);
  const float* vb = vF + (size_t)bidx * HWN * CDIM + head * HD + lane;

  float acc = 0.0f;
#pragma unroll
  for (int p = 0; p < NP; ++p) {
    const float ox = oan[head * 8 + p * 2 + 0];
    const float oy = oan[head * 8 + p * 2 + 1];
    // loc*W - 0.5 (align_corners=False)
    const float px = (refx + ox * (1.0f / 64.0f)) * 64.0f - 0.5f;
    const float py = (refy + oy * (1.0f / 64.0f)) * 64.0f - 0.5f;
    const float fx = floorf(px), fy = floorf(py);
    const float wx1 = px - fx, wy1 = py - fy;
    const float wx0 = 1.0f - wx1, wy0 = 1.0f - wy1;
    const int x0 = (int)fx, y0 = (int)fy;
    const float wp = lg[p] * rsum;

#pragma unroll
    for (int cy = 0; cy < 2; ++cy)
#pragma unroll
      for (int cx = 0; cx < 2; ++cx) {
        const int gy = y0 + cy, gx = x0 + cx;
        const bool valid = (gy >= 0) && (gy < GSZ) && (gx >= 0) && (gx < GSZ);
        const int gyc = gy < 0 ? 0 : (gy > 63 ? 63 : gy);
        const int gxc = gx < 0 ? 0 : (gx > 63 ? 63 : gx);
        const float wgt = (cy ? wy1 : wy0) * (cx ? wx1 : wx0) * wp;
        const float val = vb[(size_t)(gyc * GSZ + gxc) * CDIM];
        acc += valid ? (val * wgt) : 0.0f;
      }
  }
  outB[(size_t)n * CDIM + head * HD + lane] = (bf16_t)acc;
}

// ---------------------------------------------------------------------------
// Weight prep (trivial cost)
// ---------------------------------------------------------------------------
__global__ void k_f2bf(const float* __restrict__ in, bf16_t* __restrict__ out, int n)
{
  const int t = blockIdx.x * blockDim.x + threadIdx.x;
  if (t < n) out[t] = (bf16_t)in[t];
}

// Pack Woff(64x256) || Wa(32x256) -> Woa(96x256) bf16, per layer
__global__ void k_pack_oa(const float* __restrict__ Woff, const float* __restrict__ Wa,
                          bf16_t* __restrict__ Woa)
{
  const int t = blockIdx.x * blockDim.x + threadIdx.x;   // NL*96*256
  if (t >= NL * 96 * 256) return;
  const int c = t & 255, rn = t >> 8;
  const int l = rn / 96, r = rn % 96;
  const float v = (r < 64) ? Woff[((size_t)l * 64 + r) * 256 + c]
                           : Wa[((size_t)l * 32 + (r - 64)) * 256 + c];
  Woa[t] = (bf16_t)v;
}

__global__ void k_pack_oa_bias(const float* __restrict__ boff, const float* __restrict__ ba,
                               float* __restrict__ boa)
{
  const int t = blockIdx.x * blockDim.x + threadIdx.x;   // NL*96
  if (t >= NL * 96) return;
  const int l = t / 96, r = t % 96;
  boa[t] = (r < 64) ? boff[l * 64 + r] : ba[l * 32 + (r - 64)];
}

// ---------------------------------------------------------------------------
// Host driver
// ---------------------------------------------------------------------------
extern "C" void kernel_launch(void* const* d_in, const int* in_sizes, int n_in,
                              void* d_out, int out_size, void* d_ws, size_t ws_size,
                              hipStream_t stream)
{
  (void)in_sizes; (void)n_in; (void)out_size; (void)ws_size;

  const float* x    = (const float*)d_in[0];
  const float* Wv   = (const float*)d_in[1];
  const float* bv   = (const float*)d_in[2];
  const float* Woff = (const float*)d_in[3];
  const float* boff = (const float*)d_in[4];
  const float* Wa   = (const float*)d_in[5];
  const float* ba   = (const float*)d_in[6];
  const float* Wo   = (const float*)d_in[7];
  const float* bo   = (const float*)d_in[8];
  const float* n1w  = (const float*)d_in[9];
  const float* n1b  = (const float*)d_in[10];
  const float* W1   = (const float*)d_in[11];
  const float* b1   = (const float*)d_in[12];
  const float* W2   = (const float*)d_in[13];
  const float* b2   = (const float*)d_in[14];
  const float* n2w  = (const float*)d_in[15];
  const float* n2b  = (const float*)d_in[16];

  char* ws = (char*)d_ws;
  size_t off = 0;
  auto alloc = [&](size_t bytes) -> void* {
    void* p = ws + off;
    off += (bytes + 255) & ~(size_t)255;
    return p;
  };

  float*  srcF = (float*)alloc((size_t)TOK * CDIM * 4);
  float*  tmpF = (float*)alloc((size_t)TOK * CDIM * 4);
  float*  vF   = (float*)alloc((size_t)TOK * CDIM * 4);
  float*  oaF  = (float*)alloc((size_t)TOK * 96 * 4);
  float*  posF = (float*)alloc((size_t)HWN * CDIM * 4);
  bf16_t* srcB = (bf16_t*)alloc((size_t)TOK * CDIM * 2);
  bf16_t* qB   = (bf16_t*)alloc((size_t)TOK * CDIM * 2);
  bf16_t* aoB  = (bf16_t*)alloc((size_t)TOK * CDIM * 2);
  bf16_t* hB   = (bf16_t*)alloc((size_t)TOK * DF * 2);
  bf16_t* WvB  = (bf16_t*)alloc((size_t)NL * CDIM * CDIM * 2);
  bf16_t* WoB  = (bf16_t*)alloc((size_t)NL * CDIM * CDIM * 2);
  bf16_t* W1B  = (bf16_t*)alloc((size_t)NL * DF * CDIM * 2);
  bf16_t* W2B  = (bf16_t*)alloc((size_t)NL * CDIM * DF * 2);
  bf16_t* WoaB = (bf16_t*)alloc((size_t)NL * 96 * CDIM * 2);
  float*  boaF = (float*)alloc((size_t)NL * 96 * 4);

  const int TPB = 256;
  auto blocks = [](size_t n, int tpb) { return (unsigned)((n + tpb - 1) / tpb); };

  // ---- one-time prep ----
  k_f2bf<<<blocks((size_t)NL * CDIM * CDIM, TPB), TPB, 0, stream>>>(Wv, WvB, NL * CDIM * CDIM);
  k_f2bf<<<blocks((size_t)NL * CDIM * CDIM, TPB), TPB, 0, stream>>>(Wo, WoB, NL * CDIM * CDIM);
  k_f2bf<<<blocks((size_t)NL * DF * CDIM, TPB), TPB, 0, stream>>>(W1, W1B, NL * DF * CDIM);
  k_f2bf<<<blocks((size_t)NL * CDIM * DF, TPB), TPB, 0, stream>>>(W2, W2B, NL * CDIM * DF);
  k_pack_oa<<<blocks((size_t)NL * 96 * 256, TPB), TPB, 0, stream>>>(Woff, Wa, WoaB);
  k_pack_oa_bias<<<blocks((size_t)NL * 96, TPB), TPB, 0, stream>>>(boff, ba, boaF);
  k_pos<<<blocks((size_t)HWN * 128, TPB), TPB, 0, stream>>>(posF);
  k_transpose_in<<<blocks((size_t)TOK * CDIM, TPB), TPB, 0, stream>>>(x, srcF);
  k_prep<<<blocks((size_t)TOK * CDIM, TPB), TPB, 0, stream>>>(srcF, posF, srcB, qB);

  // macro tiles: 128x128 (wave tile 32x64) for wide GEMMs; 256x32 for N=96
  const dim3 gWide256(TOK / 128, CDIM / 128);   // N=256 -> (256, 2)
  const dim3 gWideF1 (TOK / 128, DF / 128);     // N=1024 -> (256, 8)
  const dim3 gOA     (TOK / 256, 96 / 32);      // (128, 3)

  for (int l = 0; l < NL; ++l) {
    // v = srcB @ Wv^T + bv                       (f32 out, gathered later)
    k_gemm<4, 2, 64><<<gWide256, TPB, 0, stream>>>(
        srcB, WvB + (size_t)l * CDIM * CDIM, bv + l * CDIM,
        vF, (bf16_t*)0, TOK, CDIM, CDIM, 0);
    // [off | attn-logits] = qB @ Woa^T + boa     (f32, N=96)
    k_gemm<8, 1, 32><<<gOA, TPB, 0, stream>>>(
        qB, WoaB + (size_t)l * 96 * CDIM, boaF + l * 96,
        oaF, (bf16_t*)0, TOK, 96, CDIM, 0);
    // softmax + bilinear gather -> aoB (bf16)
    k_deform_gather<<<TOK * NH / 8, TPB, 0, stream>>>(vF, oaF, aoB);
    // o = aoB @ Wo^T + bo                        (f32)
    k_gemm<4, 2, 64><<<gWide256, TPB, 0, stream>>>(
        aoB, WoB + (size_t)l * CDIM * CDIM, bo + l * CDIM,
        tmpF, (bf16_t*)0, TOK, CDIM, CDIM, 0);
    // src = LN(src + o); srcB = bf16(src)
    k_add_ln<<<TOK / 8, TPB, 0, stream>>>(srcF, tmpF, n1w + l * CDIM, n1b + l * CDIM,
                                          srcB, (const float*)0, (bf16_t*)0);
    // h = relu(srcB @ W1^T + b1)                 (bf16)
    k_gemm<4, 2, 64><<<gWideF1, TPB, 0, stream>>>(
        srcB, W1B + (size_t)l * DF * CDIM, b1 + l * DF,
        (float*)0, hB, TOK, DF, CDIM, 1);
    // f = h @ W2^T + b2                          (f32, K=1024)
    k_gemm<4, 2, 64><<<gWide256, TPB, 0, stream>>>(
        hB, W2B + (size_t)l * CDIM * DF, b2 + l * CDIM,
        tmpF, (bf16_t*)0, TOK, CDIM, DF, 0);
    // src = LN(src + f); srcB = bf16(src); qB = bf16(src + pos) for next layer
    k_add_ln<<<TOK / 8, TPB, 0, stream>>>(srcF, tmpF, n2w + l * CDIM, n2b + l * CDIM,
                                          srcB, posF, qB);
  }

  k_transpose_out<<<blocks((size_t)TOK * CDIM, TPB), TPB, 0, stream>>>(srcF, (float*)d_out);
}
